// MemoryEfficientAttention_68951404970274
// MI455X (gfx1250) — matrix operand order repaired and checked by reading, not compiled, not run on hardware
//
#include <hip/hip_runtime.h>
#include <hip/hip_bf16.h>

// ---------------------------------------------------------------------------
// MHA forward for MI455X (gfx1250, wave32, WMMA).
//   B=4, S=2048, D=512, H=8, HD=64
// Pipeline:
//   1) Q/K/V projections: X[8192x512](f32) @ W^T(f32) -> f16, head-major [B,H,S,HD]
//   2) flash attention per (b,h), 128 q-rows per block, online softmax,
//      K tiles double-buffered in LDS via global_load_async_to_lds_b128
//   3) out projection: AttO[8192x512](f16) @ Wo^T + bo -> f32
// All matrix math via v_wmma_f32_16x16x32_f16 (f16 in, f32 accum).
// ---------------------------------------------------------------------------

typedef __attribute__((ext_vector_type(16))) _Float16 v16h;
typedef __attribute__((ext_vector_type(8)))  float    v8f;

union FragH { v16h v; _Float16 h[16]; uint4 q[2]; };
union FragF { v8f  v; float    f[8];  };

__device__ __forceinline__ v8f wmma_f16(const v16h& a, const v16h& b, const v8f& c) {
  // D = A(16x32 f16) * B(32x16 f16) + C(16x16 f32)
  return __builtin_amdgcn_wmma_f32_16x16x32_f16(
      /*neg_a=*/false, a, /*neg_b=*/false, b,
      /*c_mod=*/(short)0, c, /*reuse_a=*/false, /*reuse_b=*/false);
}

constexpr int Bc = 4, Sc = 2048, Dc = 512, Hc = 8, HDc = 64;

// Attention LDS strides (halves). 160B rows keep all 16B fragment chunks aligned.
constexpr int KS = 80;
constexpr int VS = 80;
constexpr int PS = 80;
constexpr int KTILE = 64 * KS;  // one K tile in LDS (halves)

// Async copy of this thread's 32B slice of a 64x64 f16 K tile into LDS.
// gbase: uniform global base of the tile; r/c0: this thread's row / hd chunk.
// INST_OFFSET is added to BOTH the LDS and global address, so offset:0/16
// covers the two 16B halves. Tracked by ASYNCcnt (2 per wave per tile).
__device__ __forceinline__ void async_k_tile(const _Float16* gbase,
                                             const _Float16* ldsbase,
                                             int r, int c0) {
  unsigned voff = (unsigned)((r * HDc + c0) * 2);                 // global byte off
  unsigned lds0 = (unsigned)(uintptr_t)(ldsbase + r * KS + c0);   // LDS byte addr
  asm volatile("global_load_async_to_lds_b128 %0, %1, %2 offset:0"
               :: "v"(lds0), "v"(voff), "s"(gbase) : "memory");
  asm volatile("global_load_async_to_lds_b128 %0, %1, %2 offset:16"
               :: "v"(lds0), "v"(voff), "s"(gbase) : "memory");
}

// ===========================================================================
// GEMM: Out = A[M x K] @ W[N x K]^T + bias
//   A_F32:        A is fp32 (converted to f16 on load), else A is f16
//   OUT_F16_HEAD: write f16 into [B,H,S,HD] head-major layout,
//                 else write fp32 into flat [M,N]
// Block: 256 threads (8 waves), tile 128(M) x 64(N), K-step 32.
// ===========================================================================
template <bool A_F32, bool OUT_F16_HEAD>
__global__ __launch_bounds__(256) void gemm_xwt_kernel(
    const void* __restrict__ Av, const float* __restrict__ W,
    const float* __restrict__ bias, void* __restrict__ Out) {
  constexpr int K  = Dc;
  constexpr int N  = Dc;
  constexpr int XS = 48;   // LDS row stride (halves), 96B: 16B-aligned chunks
  constexpr int WS = 48;

  __shared__ _Float16 Xs[128 * XS];
  __shared__ _Float16 Ws[64 * WS];

  const int tid    = threadIdx.x;
  const int lane   = tid & 31;
  const int wid    = tid >> 5;
  const int half   = lane >> 4;
  const int nl     = lane & 15;
  const int wave_m = wid & 3;    // 0..3  -> 32-row slabs of the 128-row tile
  const int wave_n = wid >> 2;   // 0..1  -> 32-col slabs of the 64-col tile
  const int m0     = blockIdx.y * 128;
  const int n0     = blockIdx.x * 64;

  FragF acc[2][2];
#pragma unroll
  for (int mt = 0; mt < 2; mt++)
#pragma unroll
    for (int nt = 0; nt < 2; nt++) acc[mt][nt].v = (v8f){};

  for (int k0 = 0; k0 < K; k0 += 32) {
    __syncthreads();  // previous iteration's fragment reads complete

    // ---- stage A tile: 128 rows x 32 k (16 halves per thread) ----
    {
      const int r  = tid >> 1;
      const int c0 = (tid & 1) * 16;
      FragH t;
      if constexpr (A_F32) {
        const float4* ap =
            (const float4*)((const float*)Av + (size_t)(m0 + r) * K + k0 + c0);
#pragma unroll
        for (int i = 0; i < 4; i++) {
          float4 f = ap[i];
          t.h[4 * i + 0] = (_Float16)f.x;
          t.h[4 * i + 1] = (_Float16)f.y;
          t.h[4 * i + 2] = (_Float16)f.z;
          t.h[4 * i + 3] = (_Float16)f.w;
        }
      } else {
        const uint4* ap =
            (const uint4*)((const _Float16*)Av + (size_t)(m0 + r) * K + k0 + c0);
        t.q[0] = ap[0];
        t.q[1] = ap[1];
      }
      *(uint4*)&Xs[r * XS + c0]     = t.q[0];
      *(uint4*)&Xs[r * XS + c0 + 8] = t.q[1];
      if (k0 + 32 < K) {
        __builtin_prefetch((const char*)Av +
                               (A_F32 ? 4u : 2u) * ((size_t)(m0 + r) * K + k0 + 32),
                           0, 0);
      }
    }
    // ---- stage W tile: 64 rows(N) x 32 k (8 halves per thread) ----
    {
      const int r  = tid >> 2;
      const int c0 = (tid & 3) * 8;
      const float4* wp = (const float4*)(W + (size_t)(n0 + r) * K + k0 + c0);
      float4 f0 = wp[0], f1 = wp[1];
      FragH t;
      t.h[0] = (_Float16)f0.x; t.h[1] = (_Float16)f0.y;
      t.h[2] = (_Float16)f0.z; t.h[3] = (_Float16)f0.w;
      t.h[4] = (_Float16)f1.x; t.h[5] = (_Float16)f1.y;
      t.h[6] = (_Float16)f1.z; t.h[7] = (_Float16)f1.w;
      *(uint4*)&Ws[r * WS + c0] = t.q[0];
    }
    __syncthreads();

    // ---- fragments + 4 WMMAs per wave per K-step ----
    FragH a[2], b[2];
#pragma unroll
    for (int mt = 0; mt < 2; mt++) {
      const int r = wave_m * 32 + mt * 16 + nl;  // A row in this lane
      a[mt].q[0] = *(const uint4*)&Xs[r * XS + 8 * half];       // K 0..7 / 8..15
      a[mt].q[1] = *(const uint4*)&Xs[r * XS + 16 + 8 * half];  // K 16..23 / 24..31
    }
#pragma unroll
    for (int nt = 0; nt < 2; nt++) {
      const int r = wave_n * 32 + nt * 16 + nl;  // B column in this lane
      b[nt].q[0] = *(const uint4*)&Ws[r * WS + 8 * half];       // K 0..7 / 8..15
      b[nt].q[1] = *(const uint4*)&Ws[r * WS + 16 + 8 * half];
    }
#pragma unroll
    for (int mt = 0; mt < 2; mt++)
#pragma unroll
      for (int nt = 0; nt < 2; nt++)
        acc[mt][nt].v = wmma_f16(a[mt].v, b[nt].v, acc[mt][nt].v);
  }

  // ---- epilogue: bias + store ----
#pragma unroll
  for (int mt = 0; mt < 2; mt++)
#pragma unroll
    for (int nt = 0; nt < 2; nt++)
#pragma unroll
      for (int j = 0; j < 8; j++) {
        const int row = m0 + wave_m * 32 + mt * 16 + 8 * half + j;
        const int col = n0 + wave_n * 32 + nt * 16 + nl;
        const float val = acc[mt][nt].f[j] + bias[col];
        if constexpr (OUT_F16_HEAD) {
          const int bb = row >> 11;          // row / S
          const int ss = row & (Sc - 1);     // row % S
          const int hh = col >> 6;           // col / HD
          const int hd = col & (HDc - 1);    // col % HD
          ((_Float16*)Out)[(((size_t)(bb * Hc + hh) * Sc + ss) << 6) + hd] =
              (_Float16)val;
        } else {
          ((float*)Out)[(size_t)row * N + col] = val;
        }
      }
}

// ===========================================================================
// Flash attention per (b,h). Block = 256 threads (8 waves), 128 q-rows.
// Each wave owns 16 q-rows; iterates 64-row KV tiles with online softmax.
// K tiles double-buffered via async-to-LDS; V transposed through VGPRs.
// Q/K/V in f16 head-major [B,H,S,HD]; output f16 flat [B*S, D].
// ===========================================================================
__global__ __launch_bounds__(256) void flash_attn_kernel(
    const _Float16* __restrict__ Qp, const _Float16* __restrict__ Kp,
    const _Float16* __restrict__ Vp, _Float16* __restrict__ AttO) {
  __shared__ _Float16 Ks[2 * KTILE];  // double-buffered K tile [kv][hd]
  __shared__ _Float16 Vt[64 * VS];    // V tile, transposed [hd][kv]
  __shared__ _Float16 Ps[128 * PS];   // P (per-wave 16-row slabs) [q][kv]

  const int tid  = threadIdx.x;
  const int lane = tid & 31;
  const int wid  = tid >> 5;
  const int half = lane >> 4;
  const int nl   = lane & 15;
  const int bh   = blockIdx.y;          // b*H + h
  const int q0   = blockIdx.x * 128;
  const size_t base = (size_t)bh * Sc * HDc;

  const int lr = tid >> 2;        // 0..63: staging row (kv)
  const int lc = (tid & 3) * 16;  // staging hd chunk

  // ---- Q A-fragments, resident in registers; fold in 1/sqrt(HD) = 0.125 ----
  FragH qa[2];
  {
    const int qrow = q0 + wid * 16 + nl;
    const uint4* qp = (const uint4*)(Qp + base + (size_t)qrow * HDc);
#pragma unroll
    for (int g = 0; g < 2; g++) {
      qa[g].q[0] = qp[4 * g + half];      // hd = 32g + 8*half + 0..7
      qa[g].q[1] = qp[4 * g + 2 + half];  // hd = 32g + 16 + 8*half + 0..7
#pragma unroll
      for (int e = 0; e < 16; e++) qa[g].h[e] = qa[g].h[e] * (_Float16)0.125f;
    }
  }

  // Prologue: kick off async copy of K tile 0 into buffer 0 (ASYNCcnt += 2).
  async_k_tile(Kp + base, &Ks[0], lr, lc);

  FragF o[4];
#pragma unroll
  for (int t = 0; t < 4; t++) o[t].v = (v8f){};
  float rmax[8], rl[8];
#pragma unroll
  for (int j = 0; j < 8; j++) { rmax[j] = -INFINITY; rl[j] = 0.0f; }

  constexpr int NT = Sc / 64;  // 32 KV tiles
  for (int it = 0; it < NT; ++it) {
    const int kv0 = it * 64;
    const int cur = it & 1;
    const int nxt = cur ^ 1;

    // ---- start V tile global loads early (hidden under the barrier) ----
    FragH tv;
    {
      const uint4* vp =
          (const uint4*)(Vp + base + (size_t)(kv0 + lr) * HDc + lc);
      tv.q[0] = vp[0];
      tv.q[1] = vp[1];
      if (kv0 + 64 < Sc)
        __builtin_prefetch(Vp + base + (size_t)(kv0 + 64 + lr) * HDc + lc, 0, 0);
    }

    __syncthreads();  // all waves done reading Ks[nxt], Vt, Ps of iter it-1

    // ---- issue async copy of the NEXT K tile into the other buffer ----
    if (it + 1 < NT)
      async_k_tile(Kp + base + (size_t)(kv0 + 64) * HDc, &Ks[nxt * KTILE], lr, lc);

    // ---- scatter V (transpose) into Vt ----
#pragma unroll
    for (int i = 0; i < 16; i++) Vt[(lc + i) * VS + lr] = tv.h[i];

    // ---- wait for THIS tile's K data (leave next tile's copies in flight) ----
    if (it + 1 < NT) asm volatile("s_wait_asynccnt 0x2" ::: "memory");
    else             asm volatile("s_wait_asynccnt 0x0" ::: "memory");
    __syncthreads();  // K tile + Vt visible to all waves

    const _Float16* Kc = &Ks[cur * KTILE];

    // ---- S = (Q*scale) @ K^T : 4 tiles of 16x16, K-dim 64 = 2 WMMAs each ----
    FragF sv[4];
#pragma unroll
    for (int nt = 0; nt < 4; nt++) {
      FragH b0, b1;
      const int kr = nt * 16 + nl;  // kv column owned by this lane
      b0.q[0] = *(const uint4*)&Kc[kr * KS + 8 * half];         // hd 0..31
      b0.q[1] = *(const uint4*)&Kc[kr * KS + 16 + 8 * half];
      b1.q[0] = *(const uint4*)&Kc[kr * KS + 32 + 8 * half];    // hd 32..63
      b1.q[1] = *(const uint4*)&Kc[kr * KS + 48 + 8 * half];
      v8f c = (v8f){};
      c = wmma_f16(qa[0].v, b0.v, c);
      c = wmma_f16(qa[1].v, b1.v, c);
      sv[nt].v = c;
    }

    // ---- online softmax over this tile's 64 columns ----
    float alpha[8];
#pragma unroll
    for (int j = 0; j < 8; j++) {
      float m = sv[0].f[j];
#pragma unroll
      for (int nt = 1; nt < 4; nt++) m = fmaxf(m, sv[nt].f[j]);
#pragma unroll
      for (int off = 1; off < 16; off <<= 1) m = fmaxf(m, __shfl_xor(m, off));
      const float nmax = fmaxf(rmax[j], m);
      alpha[j] = __expf(rmax[j] - nmax);
      rmax[j]  = nmax;
      float s = 0.0f;
#pragma unroll
      for (int nt = 0; nt < 4; nt++) {
        const float p = __expf(sv[nt].f[j] - nmax);
        sv[nt].f[j] = p;
        s += p;
      }
#pragma unroll
      for (int off = 1; off < 16; off <<= 1) s += __shfl_xor(s, off);
      rl[j] = alpha[j] * rl[j] + s;
    }
#pragma unroll
    for (int t = 0; t < 4; t++)
#pragma unroll
      for (int j = 0; j < 8; j++) o[t].f[j] *= alpha[j];

    // ---- C-layout P -> LDS -> A-layout fragments ----
    const int prow = wid * 16 + 8 * half;
#pragma unroll
    for (int nt = 0; nt < 4; nt++)
#pragma unroll
      for (int j = 0; j < 8; j++)
        Ps[(prow + j) * PS + nt * 16 + nl] = (_Float16)sv[nt].f[j];

    FragH pa[2];
    {
      const int r = wid * 16 + nl;
      pa[0].q[0] = *(const uint4*)&Ps[r * PS + 8 * half];        // kv 0..31
      pa[0].q[1] = *(const uint4*)&Ps[r * PS + 16 + 8 * half];
      pa[1].q[0] = *(const uint4*)&Ps[r * PS + 32 + 8 * half];   // kv 32..63
      pa[1].q[1] = *(const uint4*)&Ps[r * PS + 48 + 8 * half];
    }

    // ---- O += P @ V : 4 hd-tiles x 2 K-chunks ----
#pragma unroll
    for (int nt = 0; nt < 4; nt++) {
      FragH b0, b1;
      const int hd = nt * 16 + nl;  // hd column owned by this lane
      b0.q[0] = *(const uint4*)&Vt[hd * VS + 8 * half];         // kv 0..31
      b0.q[1] = *(const uint4*)&Vt[hd * VS + 16 + 8 * half];
      b1.q[0] = *(const uint4*)&Vt[hd * VS + 32 + 8 * half];    // kv 32..63
      b1.q[1] = *(const uint4*)&Vt[hd * VS + 48 + 8 * half];
      o[nt].v = wmma_f16(pa[0].v, b0.v, o[nt].v);
      o[nt].v = wmma_f16(pa[1].v, b1.v, o[nt].v);
    }
  }

  // ---- normalize and store to [B*S, D] f16 ----
  const int bb = bh / Hc, hh = bh % Hc;
#pragma unroll
  for (int j = 0; j < 8; j++) {
    const float inv = 1.0f / rl[j];
    const int s = q0 + wid * 16 + 8 * half + j;
    _Float16* orow = AttO + (size_t)(bb * Sc + s) * Dc + hh * HDc;
#pragma unroll
    for (int nt = 0; nt < 4; nt++)
      orow[nt * 16 + nl] = (_Float16)(o[nt].f[j] * inv);
  }
}

// ===========================================================================
extern "C" void kernel_launch(void* const* d_in, const int* in_sizes, int n_in,
                              void* d_out, int out_size, void* d_ws,
                              size_t ws_size, hipStream_t stream) {
  const float* query = (const float*)d_in[0];
  const float* key   = (const float*)d_in[1];
  const float* value = (const float*)d_in[2];
  const float* Wq    = (const float*)d_in[3];
  const float* bq    = (const float*)d_in[4];
  const float* Wk    = (const float*)d_in[5];
  const float* bk    = (const float*)d_in[6];
  const float* Wv    = (const float*)d_in[7];
  const float* bv    = (const float*)d_in[8];
  const float* Wo    = (const float*)d_in[9];
  const float* bo    = (const float*)d_in[10];
  float* out = (float*)d_out;

  const size_t NTOK = (size_t)Bc * Sc * Dc;  // 4,194,304 elements
  _Float16* Qp   = (_Float16*)d_ws;          // [B,H,S,HD] f16
  _Float16* Kp   = Qp + NTOK;
  _Float16* Vp   = Kp + NTOK;
  _Float16* AttO = Vp + NTOK;                // [B*S, D] f16 (needs 32 MB ws)

  dim3 block(256);
  dim3 ggrid(Dc / 64, (Bc * Sc) / 128);      // (8, 64)

  gemm_xwt_kernel<true, true><<<ggrid, block, 0, stream>>>(query, Wq, bq, Qp);
  gemm_xwt_kernel<true, true><<<ggrid, block, 0, stream>>>(key,   Wk, bk, Kp);
  gemm_xwt_kernel<true, true><<<ggrid, block, 0, stream>>>(value, Wv, bv, Vp);

  dim3 agrid(Sc / 128, Bc * Hc);             // (16, 32)
  flash_attn_kernel<<<agrid, block, 0, stream>>>(Qp, Kp, Vp, AttO);

  gemm_xwt_kernel<false, false><<<ggrid, block, 0, stream>>>(AttO, Wo, bo, out);
}
